// Attention_65807488909554
// MI455X (gfx1250) — matrix operand order repaired and checked by reading, not compiled, not run on hardware
//
#include <hip/hip_runtime.h>
#include <hip/hip_bf16.h>

typedef __attribute__((ext_vector_type(16))) __bf16 v16bf;
typedef __attribute__((ext_vector_type(8)))  __bf16 v8bf;
typedef __attribute__((ext_vector_type(4)))  __bf16 v4bf;
typedef __attribute__((ext_vector_type(8)))  float  v8f;

#define BATCH 2
#define SEQ   2048
#define HID_  1024
#define NH    16
#define NKVH  4
#define HD    64
#define MTOT  (BATCH * SEQ)   // 4096

__device__ __forceinline__ v16bf cat8(v8bf lo, v8bf hi) {
    return __builtin_shufflevector(lo, hi, 0,1,2,3,4,5,6,7,8,9,10,11,12,13,14,15);
}

__device__ __forceinline__ v8f wmma_bf16(v16bf a, v16bf b, v8f c) {
    return __builtin_amdgcn_wmma_f32_16x16x32_bf16(false, a, false, b, (short)0, c, false, false);
}

// ---------------------------------------------------------------------------
// Kernel 0: swizzle an f32 weight (1024 x N, row-major) into WMMA B-fragment
// order as bf16:  ws[ntile][kstep][lane][i] = w[(kstep*32 + (lane>>4)*16 + i)*N
//                                              + ntile*16 + (lane&15)]
// ---------------------------------------------------------------------------
__global__ void weight_swizzle_kernel(const float* __restrict__ w,
                                      __bf16* __restrict__ wsw, int N)
{
    int t = blockIdx.x * 256 + threadIdx.x;
    int total = (N / 16) * 32 * 32;
    if (t >= total) return;
    int lane  = t & 31;
    int kstep = (t >> 5) & 31;
    int ntile = t >> 10;
    int n     = ntile * 16 + (lane & 15);
    int kbase = kstep * 32 + (lane >> 4) * 16;
    v16bf frag;
    #pragma unroll
    for (int i = 0; i < 16; ++i)
        frag[i] = (__bf16)w[(kbase + i) * N + n];
    ((v16bf*)wsw)[t] = frag;
}

// ---------------------------------------------------------------------------
// Kernel 1: fused QKV projection. Block tile 128x64 (4 waves, 32 rows/wave so
// each B-frag feeds two WMMAs). A staged via LDS; B-frags streamed from
// pre-swizzled global weights. K/V written directly in attention frag order.
// ---------------------------------------------------------------------------
__global__ __launch_bounds__(128) void qkv_gemm_kernel(
    const float*  __restrict__ x,     // (4096,1024)
    const __bf16* __restrict__ wsq,
    const __bf16* __restrict__ wsk,
    const __bf16* __restrict__ wsv,
    __bf16* __restrict__ qo,          // (4096,1024) row-major
    __bf16* __restrict__ kw,          // QK^T B-frag order
    __bf16* __restrict__ vw)          // PV  B-frag order
{
    __shared__ __bf16 As[128][40];

    const int tid  = threadIdx.x;
    const int wave = tid >> 5;
    const int lane = tid & 31;
    const int hi   = lane >> 4;
    const int ln   = lane & 15;
    const int mblk = blockIdx.y * 128;
    const int nblk = blockIdx.x * 64;   // fused N = 1536

    const __bf16* wsB; int ncol0, mode;
    if (nblk < 1024)      { wsB = wsq; ncol0 = nblk;        mode = 0; }
    else if (nblk < 1280) { wsB = wsk; ncol0 = nblk - 1024; mode = 1; }
    else                  { wsB = wsv; ncol0 = nblk - 1280; mode = 2; }
    const v16bf* Bf = (const v16bf*)wsB;
    const int ntile0 = ncol0 >> 4;

    v8f acc0[4] = {}, acc1[4] = {};
    for (int ks = 0; ks < 32; ++ks) {
        // stage A tile 128x32 (f32 -> bf16); uniform trip count, no divergence
        #pragma unroll
        for (int it = 0; it < 8; ++it) {
            int t = tid + it * 128;
            int r = t >> 3, c4 = (t & 7) * 4;
            const float4 f = *(const float4*)&x[(mblk + r) * HID_ + ks * 32 + c4];
            v4bf hh; hh[0] = (__bf16)f.x; hh[1] = (__bf16)f.y;
                     hh[2] = (__bf16)f.z; hh[3] = (__bf16)f.w;
            *(v4bf*)&As[r][c4] = hh;
            if (ks < 31)
                __builtin_prefetch(&x[(mblk + r) * HID_ + (ks + 1) * 32 + c4], 0, 3);
        }
        __syncthreads();

        const int arow0 = wave * 32 + ln;
        const int arow1 = arow0 + 16;
        v16bf a0 = cat8(*(const v8bf*)&As[arow0][16 * hi],
                        *(const v8bf*)&As[arow0][16 * hi + 8]);
        v16bf a1 = cat8(*(const v8bf*)&As[arow1][16 * hi],
                        *(const v8bf*)&As[arow1][16 * hi + 8]);
        #pragma unroll
        for (int nt = 0; nt < 4; ++nt) {
            v16bf bfr = Bf[((ntile0 + nt) * 32 + ks) * 32 + lane];
            acc0[nt] = wmma_bf16(a0, bfr, acc0[nt]);
            acc1[nt] = wmma_bf16(a1, bfr, acc1[nt]);
        }
        __syncthreads();
    }

    // epilogue (both row groups)
    #pragma unroll
    for (int rg = 0; rg < 2; ++rg) {
        v8f* accp = rg ? acc1 : acc0;
        #pragma unroll
        for (int nt = 0; nt < 4; ++nt)
            #pragma unroll
            for (int r = 0; r < 8; ++r) {
                int m = mblk + wave * 32 + rg * 16 + r + 8 * hi;
                __bf16 val = (__bf16)accp[nt][r];
                if (mode == 0) {
                    qo[m * 1024 + ncol0 + nt * 16 + ln] = val;
                } else {
                    int b = m >> 11, s = m & 2047;
                    int dcol = ncol0 + nt * 16 + ln;     // 0..255
                    int kvh = dcol >> 6, drem = dcol & 63;
                    int kt = s >> 5;
                    if (mode == 1) {   // K: ln=key, (hi,i)=d
                        int db = drem >> 5, hk = (drem >> 4) & 1, ik = drem & 15;
                        int nt2 = (s >> 4) & 1, lk = s & 15;
                        size_t idx = ((size_t)((((b * NKVH + kvh) * 64 + kt) * 2 + nt2) * 2 + db)) * 512
                                   + (hk * 16 + lk) * 16 + ik;
                        kw[idx] = val;
                    } else {           // V: ln=d, (hi,i)=key
                        int ntv = drem >> 4, lv = drem & 15;
                        int hv = (s >> 4) & 1, iv = s & 15;
                        size_t idx = ((size_t)(((b * NKVH + kvh) * 64 + kt) * 4 + ntv)) * 512
                                   + (hv * 16 + lv) * 16 + iv;
                        vw[idx] = val;
                    }
                }
            }
    }
}

// ---------------------------------------------------------------------------
// Kernel 2: RoPE. q row-major (folds 1/sqrt(D)); k in swizzled QK^T frag
// layout where the (d, d+32) pair sits 512 halves apart.
// ---------------------------------------------------------------------------
__global__ void rope_kernel(__bf16* __restrict__ q, __bf16* __restrict__ kw,
                            const float* __restrict__ cosb,
                            const float* __restrict__ sinb)
{
    const int qpairs = MTOT * 1024 / 2;
    const int kpairs = MTOT * 256 / 2;
    int idx = blockIdx.x * 256 + threadIdx.x;
    if (idx < qpairs) {
        int m = idx >> 9;
        int pr = idx & 511;
        int head = pr >> 5, d = pr & 31;
        int s = m & 2047;
        int c0 = m * 1024 + head * 64 + d;
        float a0 = (float)q[c0], a1 = (float)q[c0 + 32];
        float cs0 = cosb[s * 64 + d],      sn0 = sinb[s * 64 + d];
        float cs1 = cosb[s * 64 + d + 32], sn1 = sinb[s * 64 + d + 32];
        q[c0]      = (__bf16)((a0 * cs0 - a1 * sn0) * 0.125f);
        q[c0 + 32] = (__bf16)((a1 * cs1 + a0 * sn1) * 0.125f);
    } else if (idx < qpairs + kpairs) {
        int li  = idx - qpairs;      // b(1) kvh(2) kt(6) nt2(1) hk(1) lk(4) i(4)
        int i   = li & 15;
        int lk  = (li >> 4) & 15;
        int hk  = (li >> 8) & 1;
        int nt2 = (li >> 9) & 1;
        int kt  = (li >> 10) & 63;
        int kvh = (li >> 16) & 3;
        int b   = li >> 18;
        int d   = hk * 16 + i;
        int s   = kt * 32 + nt2 * 16 + lk;
        size_t i0 = ((size_t)((((b * NKVH + kvh) * 64 + kt) * 2 + nt2) * 2)) * 512
                  + (hk * 16 + lk) * 16 + i;
        size_t i1 = i0 + 512;
        float a0 = (float)kw[i0], a1 = (float)kw[i1];
        float cs0 = cosb[s * 64 + d],      sn0 = sinb[s * 64 + d];
        float cs1 = cosb[s * 64 + d + 32], sn1 = sinb[s * 64 + d + 32];
        kw[i0] = (__bf16)(a0 * cs0 - a1 * sn0);
        kw[i1] = (__bf16)(a1 * cs1 + a0 * sn1);
    }
}

// ---------------------------------------------------------------------------
// Kernel 3: causal flash attention. 8 independent waves per block, each owns
// a 32-row q tile (two row-tiles sharing every K/V fragment -> 16 WMMAs per
// 32-key tile). Only LDS use is the per-wave P->A-frag transpose. No barriers.
// ---------------------------------------------------------------------------
__global__ __launch_bounds__(256) void flash_kernel(
    const __bf16* __restrict__ q,    // (4096,1024) rope'd + prescaled
    const __bf16* __restrict__ kw,   // QK^T frag layout, rope'd
    const __bf16* __restrict__ vw,   // PV frag layout
    __bf16* __restrict__ o)          // (4096,1024)
{
    __shared__ __bf16 Ps[8][32][40];

    const int tid  = threadIdx.x;
    const int wave = tid >> 5;
    const int lane = tid & 31;
    const int hi   = lane >> 4;
    const int ln   = lane & 15;
    const int h    = blockIdx.y;
    const int b    = blockIdx.z;
    const int kvh  = h >> 2;
    const int q0   = blockIdx.x * 256 + wave * 32;

    // Q A-frags straight from global: 2 row-tiles x 2 d-blocks
    v16bf aq[2][2];
    #pragma unroll
    for (int rt = 0; rt < 2; ++rt) {
        const __bf16* qrow = q + (size_t)(b * SEQ + q0 + rt * 16 + ln) * 1024 + h * HD;
        aq[rt][0] = cat8(*(const v8bf*)(qrow + 16 * hi),
                         *(const v8bf*)(qrow + 16 * hi + 8));
        aq[rt][1] = cat8(*(const v8bf*)(qrow + 32 + 16 * hi),
                         *(const v8bf*)(qrow + 40 + 16 * hi));
    }

    float mrow[2][8], lrow[2][8];
    v8f oacc[2][4] = {};
    #pragma unroll
    for (int rt = 0; rt < 2; ++rt)
        #pragma unroll
        for (int r = 0; r < 8; ++r) { mrow[rt][r] = -1e30f; lrow[rt][r] = 0.0f; }

    const v16bf* Kf = (const v16bf*)kw + (size_t)((b * NKVH + kvh) * 64) * 128;
    const v16bf* Vf = (const v16bf*)vw + (size_t)((b * NKVH + kvh) * 64) * 128;
    const int nkt = ((q0 + 31) >> 5) + 1;

    for (int kt = 0; kt < nkt; ++kt) {
        const v16bf* Kt = Kf + kt * 128;
        v16bf k00 = Kt[lane],      k01 = Kt[32 + lane];
        v16bf k10 = Kt[64 + lane], k11 = Kt[96 + lane];

        v8f s[2][2] = {};
        #pragma unroll
        for (int rt = 0; rt < 2; ++rt) {
            s[rt][0] = wmma_bf16(aq[rt][0], k00, s[rt][0]);
            s[rt][0] = wmma_bf16(aq[rt][1], k01, s[rt][0]);
            s[rt][1] = wmma_bf16(aq[rt][0], k10, s[rt][1]);
            s[rt][1] = wmma_bf16(aq[rt][1], k11, s[rt][1]);
        }

        const int kj0 = kt * 32 + ln;
        const int kj1 = kj0 + 16;
        #pragma unroll
        for (int rt = 0; rt < 2; ++rt) {
            #pragma unroll
            for (int r = 0; r < 8; ++r) {
                int qi = q0 + rt * 16 + r + 8 * hi;
                float v0 = s[rt][0][r], v1 = s[rt][1][r];
                if (kj0 > qi) v0 = -1e30f;
                if (kj1 > qi) v1 = -1e30f;
                float lm = fmaxf(v0, v1);
                lm = fmaxf(lm, __shfl_xor(lm, 1));
                lm = fmaxf(lm, __shfl_xor(lm, 2));
                lm = fmaxf(lm, __shfl_xor(lm, 4));
                lm = fmaxf(lm, __shfl_xor(lm, 8));
                float mnew = fmaxf(mrow[rt][r], lm);
                float corr = __expf(mrow[rt][r] - mnew);
                v0 = __expf(v0 - mnew);
                v1 = __expf(v1 - mnew);
                float rs = v0 + v1;
                rs += __shfl_xor(rs, 1);
                rs += __shfl_xor(rs, 2);
                rs += __shfl_xor(rs, 4);
                rs += __shfl_xor(rs, 8);
                lrow[rt][r] = lrow[rt][r] * corr + rs;
                mrow[rt][r] = mnew;
                #pragma unroll
                for (int nt = 0; nt < 4; ++nt) oacc[rt][nt][r] *= corr;
                Ps[wave][rt * 16 + r + 8 * hi][ln]      = (__bf16)v0;
                Ps[wave][rt * 16 + r + 8 * hi][16 + ln] = (__bf16)v1;
            }
        }

        v16bf ap[2];
        #pragma unroll
        for (int rt = 0; rt < 2; ++rt)
            ap[rt] = cat8(*(const v8bf*)&Ps[wave][rt * 16 + ln][16 * hi],
                          *(const v8bf*)&Ps[wave][rt * 16 + ln][16 * hi + 8]);

        const v16bf* Vt = Vf + kt * 128;
        #pragma unroll
        for (int nt = 0; nt < 4; ++nt) {
            v16bf vf = Vt[nt * 32 + lane];
            oacc[0][nt] = wmma_bf16(ap[0], vf, oacc[0][nt]);
            oacc[1][nt] = wmma_bf16(ap[1], vf, oacc[1][nt]);
        }
    }

    #pragma unroll
    for (int rt = 0; rt < 2; ++rt) {
        float inv[8];
        #pragma unroll
        for (int r = 0; r < 8; ++r) inv[r] = 1.0f / lrow[rt][r];
        #pragma unroll
        for (int nt = 0; nt < 4; ++nt)
            #pragma unroll
            for (int r = 0; r < 8; ++r) {
                int m = b * SEQ + q0 + rt * 16 + r + 8 * hi;
                o[(size_t)m * 1024 + h * HD + nt * 16 + ln] =
                    (__bf16)(oacc[rt][nt][r] * inv[r]);
            }
    }
}

// ---------------------------------------------------------------------------
// Kernel 4: output projection. Block tile 128x64, 32 rows/wave.
// ---------------------------------------------------------------------------
__global__ __launch_bounds__(128) void out_gemm_kernel(
    const __bf16* __restrict__ a,     // (4096,1024)
    const __bf16* __restrict__ wso,   // swizzled (64 ntiles)
    float* __restrict__ out)          // (4096,1024) f32
{
    __shared__ __bf16 As[128][40];

    const int tid  = threadIdx.x;
    const int wave = tid >> 5;
    const int lane = tid & 31;
    const int hi   = lane >> 4;
    const int ln   = lane & 15;
    const int mblk = blockIdx.y * 128;
    const int nblk = blockIdx.x * 64;
    const v16bf* Bf = (const v16bf*)wso;
    const int ntile0 = nblk >> 4;

    v8f acc0[4] = {}, acc1[4] = {};
    for (int ks = 0; ks < 32; ++ks) {
        #pragma unroll
        for (int it = 0; it < 4; ++it) {
            int t = tid + it * 128;
            int r = t >> 2, c8 = (t & 3) * 8;
            *(v8bf*)&As[r][c8] =
                *(const v8bf*)&a[(size_t)(mblk + r) * HID_ + ks * 32 + c8];
        }
        __syncthreads();

        const int arow0 = wave * 32 + ln;
        const int arow1 = arow0 + 16;
        v16bf a0 = cat8(*(const v8bf*)&As[arow0][16 * hi],
                        *(const v8bf*)&As[arow0][16 * hi + 8]);
        v16bf a1 = cat8(*(const v8bf*)&As[arow1][16 * hi],
                        *(const v8bf*)&As[arow1][16 * hi + 8]);
        #pragma unroll
        for (int nt = 0; nt < 4; ++nt) {
            v16bf bfr = Bf[((ntile0 + nt) * 32 + ks) * 32 + lane];
            acc0[nt] = wmma_bf16(a0, bfr, acc0[nt]);
            acc1[nt] = wmma_bf16(a1, bfr, acc1[nt]);
        }
        __syncthreads();
    }

    #pragma unroll
    for (int rg = 0; rg < 2; ++rg) {
        v8f* accp = rg ? acc1 : acc0;
        #pragma unroll
        for (int nt = 0; nt < 4; ++nt)
            #pragma unroll
            for (int r = 0; r < 8; ++r) {
                int m = mblk + wave * 32 + rg * 16 + r + 8 * hi;
                out[(size_t)m * HID_ + nblk + nt * 16 + ln] = accp[nt][r];
            }
    }
}

// ---------------------------------------------------------------------------
extern "C" void kernel_launch(void* const* d_in, const int* in_sizes, int n_in,
                              void* d_out, int out_size, void* d_ws, size_t ws_size,
                              hipStream_t stream)
{
    const float* x    = (const float*)d_in[0];
    const float* cosb = (const float*)d_in[1];
    const float* sinb = (const float*)d_in[2];
    const float* wq   = (const float*)d_in[3];
    const float* wk   = (const float*)d_in[4];
    const float* wv   = (const float*)d_in[5];
    const float* wo   = (const float*)d_in[6];
    float* out = (float*)d_out;

    char* ws = (char*)d_ws;
    const size_t MB = 1024 * 1024;
    __bf16* qb  = (__bf16*)(ws);                 // 8 MB   row-major q
    __bf16* kwv = (__bf16*)(ws + 8  * MB);       // 2 MB   K frag layout
    __bf16* vwv = (__bf16*)(ws + 10 * MB);       // 2 MB   V frag layout
    __bf16* ab  = (__bf16*)(ws + 12 * MB);       // 8 MB   attention out
    __bf16* wsq = (__bf16*)(ws + 20 * MB);       // 2 MB   swizzled wq
    __bf16* wsk = (__bf16*)(ws + 22 * MB);       // 0.5 MB swizzled wk
    __bf16* wsv = (__bf16*)(ws + 2253 * 1024);   // 0.5 MB swizzled wv
    __bf16* wso = (__bf16*)(ws + 23 * MB);       // 2 MB   swizzled wo

    weight_swizzle_kernel<<<(65536 + 255) / 256, 256, 0, stream>>>(wq, wsq, 1024);
    weight_swizzle_kernel<<<(16384 + 255) / 256, 256, 0, stream>>>(wk, wsk, 256);
    weight_swizzle_kernel<<<(16384 + 255) / 256, 256, 0, stream>>>(wv, wsv, 256);
    weight_swizzle_kernel<<<(65536 + 255) / 256, 256, 0, stream>>>(wo, wso, 1024);

    dim3 g1(24, 32, 1);   // N = 1536 fused, M = 4096 / 128
    qkv_gemm_kernel<<<g1, 128, 0, stream>>>(x, wsq, wsk, wsv, qb, kwv, vwv);

    int pairs = MTOT * 1024 / 2 + MTOT * 256 / 2;
    rope_kernel<<<(pairs + 255) / 256, 256, 0, stream>>>(qb, kwv, cosb, sinb);

    dim3 g3(SEQ / 256, NH, BATCH);
    flash_kernel<<<g3, 256, 0, stream>>>(qb, kwv, vwv, ab);

    dim3 g4(16, 32, 1);   // N = 1024, M = 4096 / 128
    out_gemm_kernel<<<g4, 128, 0, stream>>>(ab, wso, out);
}
